// SwinTransformerBlock_60344290509067
// MI455X (gfx1250) — matrix operand order repaired and checked by reading, not compiled, not run on hardware
//
#include <hip/hip_runtime.h>
#include <hip/hip_bf16.h>
#include <math.h>
#include <stdint.h>

typedef __attribute__((ext_vector_type(16))) _Float16 v16h;
typedef __attribute__((ext_vector_type(8)))  float    v8f;
typedef __attribute__((ext_vector_type(4)))  int      v4i;

#if __has_builtin(__builtin_amdgcn_global_load_async_to_lds_b128) && \
    __has_builtin(__builtin_amdgcn_s_wait_asynccnt)
#define USE_ASYNC_LDS 1
#else
#define USE_ASYNC_LDS 0
#endif

union FragU { v16h v; unsigned u[8]; };

__device__ inline v8f vzero8() {
  v8f z;
#pragma unroll
  for (int i = 0; i < 8; ++i) z[i] = 0.0f;
  return z;
}

__device__ inline v8f wmma32(v16h a, v16h b, v8f c) {
  // D = A(16x32 f16) * B(32x16 f16) + C(16x16 f32)
  return __builtin_amdgcn_wmma_f32_16x16x32_f16(false, a, false, b, (short)0, c, false, false);
}

// A fragment: rows [row0..row0+15] of a row-major f16 tile, row stride ldh halves, K offset k0.
// ISA 16-bit A layout: lane m (0..15)/h group; VGPR q holds K pair (q&3)*2 + (q>>2)*16 + h*8.
__device__ inline v16h load_afrag(const _Float16* tile, int row0, int ldh, int k0, int lane) {
  const int h = (lane >> 4) & 1, m = lane & 15;
  const unsigned* p = (const unsigned*)(tile + (size_t)(row0 + m) * ldh + k0);
  FragU f;
#pragma unroll
  for (int q = 0; q < 8; ++q) {
    const int kk = (q & 3) * 2 + (q >> 2) * 16 + h * 8;
    f.u[q] = p[kk >> 1];
  }
  return f.v;
}

// B fragment from transposed layout BT[n][k]: cols n0..n0+15, row stride ldh halves, K offset k0.
// B layout matches A: VGPR q holds K pair (q&3)*2 + (q>>2)*16 + h*8.
__device__ inline v16h load_bfrag(const _Float16* bt, int n0, int ldh, int k0, int lane) {
  const int h = (lane >> 4) & 1, n = lane & 15;
  const unsigned* p = (const unsigned*)(bt + (size_t)(n0 + n) * ldh + k0);
  FragU f;
#pragma unroll
  for (int q = 0; q < 8; ++q) f.u[q] = p[((q & 3) * 2 + (q >> 2) * 16 + h * 8) >> 1];
  return f.v;
}

// ---- 16B global -> LDS copy: CDNA5 async-to-LDS if available, else reg staging ----
__device__ __forceinline__ void cp16(const _Float16* g, _Float16* l) {
#if USE_ASYNC_LDS
  __builtin_amdgcn_global_load_async_to_lds_b128(
      (__attribute__((address_space(1))) v4i*)(unsigned long long)(uintptr_t)g,
      (__attribute__((address_space(3))) v4i*)(unsigned)(uintptr_t)l, 0, 0);
#else
  *(uint4*)l = *(const uint4*)g;
#endif
}

__device__ __forceinline__ void stage32(const _Float16* g, _Float16* l) {
  cp16(g, l);
  cp16(g + 8, l + 8);
}

// ---------------- 128x64 GEMM core, double-buffered LDS, async prefetch ----------------
// a0/a1: per-thread global base for A chunk rows (t>>1) and 64+(t>>1) (incl. part*16);
// aStepX: 1 to advance with k0, 0 for the zero-pad row. b0: BT row base (incl. part*16).
__device__ __forceinline__ void gemm_core_128x64(
    const _Float16* a0, int aStep0, const _Float16* a1, int aStep1,
    const _Float16* b0, int K,
    _Float16* sA /*2*128*32*/, _Float16* sB /*2*64*32*/,
    v8f accLo[4], v8f accHi[4], int t) {
  const int lane = t & 31, wv = t >> 5;
  const int r = t >> 1, part = t & 1;
  auto prefetch = [&](int buf, int k0) {
    _Float16* lA0 = sA + buf * 4096 + r * 32 + part * 16;
    _Float16* lA1 = sA + buf * 4096 + (64 + r) * 32 + part * 16;
    _Float16* lB  = sB + buf * 2048 + r * 32 + part * 16;
    stage32(a0 + aStep0 * k0, lA0);
    stage32(a1 + aStep1 * k0, lA1);
    stage32(b0 + k0, lB);
  };
  const int nk = K >> 5;
  prefetch(0, 0);
  for (int kt = 0; kt < nk; ++kt) {
    const int cur = kt & 1;
#if USE_ASYNC_LDS
    __builtin_amdgcn_s_wait_asynccnt(0);
#endif
    __syncthreads();
    if (kt + 1 < nk) prefetch(cur ^ 1, (kt + 1) * 32);
    const _Float16* cA = sA + cur * 4096;
    const _Float16* cB = sB + cur * 2048;
    const v16h fa0 = load_afrag(cA, wv * 32, 32, 0, lane);
    const v16h fa1 = load_afrag(cA, wv * 32 + 16, 32, 0, lane);
#pragma unroll
    for (int j = 0; j < 4; ++j) {
      const v16h fb = load_bfrag(cB, j * 16, 32, 0, lane);
      accLo[j] = wmma32(fa0, fb, accLo[j]);
      accHi[j] = wmma32(fa1, fb, accHi[j]);
    }
    __syncthreads();
  }
}

// ---------------- weight transpose + f16 convert: dst[n*K+k] = src[k*N+n] ----------------
__global__ __launch_bounds__(256) void transpose_w_kernel(const float* __restrict__ src,
                                                          _Float16* __restrict__ dst,
                                                          int K, int N) {
  const int idx = blockIdx.x * 256 + threadIdx.x;
  if (idx < K * N) {
    const int k = idx / N, n = idx - k * N;
    dst[(size_t)n * K + k] = (_Float16)src[idx];
  }
}

__global__ __launch_bounds__(32) void zero_pad_kernel(_Float16* z) {
  z[threadIdx.x] = (_Float16)0.0f;
}

// ---------------- block reduction over 128 threads ----------------
__device__ inline float block_sum128(float v, float* sh, int t) {
  sh[t] = v;
  __syncthreads();
  if (t < 64) sh[t] += sh[t + 64];
  __syncthreads();
  if (t < 32) {
    float a = sh[t] + sh[t + 32];
#pragma unroll
    for (int o = 16; o > 0; o >>= 1) a += __shfl_down(a, o, 32);
    if (t == 0) sh[0] = a;
  }
  __syncthreads();
  const float r = sh[0];
  __syncthreads();
  return r;
}

// ---------------- LN1: x (f32) -> xn (f16) ----------------
__global__ __launch_bounds__(128) void ln1_kernel(const float* __restrict__ x,
    const float* __restrict__ g, const float* __restrict__ b, _Float16* __restrict__ xn) {
  __shared__ float sh[128];
  const size_t tk = blockIdx.x;
  const int t = threadIdx.x;
  const float* xp = x + tk * 384;
  const float v0 = xp[t], v1 = xp[t + 128], v2 = xp[t + 256];
  const float s  = block_sum128(v0 + v1 + v2, sh, t);
  const float s2 = block_sum128(v0 * v0 + v1 * v1 + v2 * v2, sh, t);
  const float mu = s * (1.0f / 384.0f);
  const float var = s2 * (1.0f / 384.0f) - mu * mu;
  const float rs = rsqrtf(var + 1e-5f);
  _Float16* op = xn + tk * 384;
  op[t]       = (_Float16)((v0 - mu) * rs * g[t]       + b[t]);
  op[t + 128] = (_Float16)((v1 - mu) * rs * g[t + 128] + b[t + 128]);
  op[t + 256] = (_Float16)((v2 - mu) * rs * g[t + 256] + b[t + 256]);
}

// ---------------- QKV GEMM with shifted-window gather of A ----------------
// grid (18, 1024); each block: 128 padded rows (2 windows) x 64 output cols.
__global__ __launch_bounds__(128) void gemm_qkv_kernel(const _Float16* __restrict__ xn,
    const _Float16* __restrict__ zbuf,
    const _Float16* __restrict__ wT, const float* __restrict__ bias,
    _Float16* __restrict__ qkv) {
  __shared__ _Float16 sA[2 * 128 * 32];
  __shared__ _Float16 sB[2 * 64 * 32];
  const int t = threadIdx.x, lane = t & 31, wv = t >> 5;
  const int row0 = blockIdx.y * 128;
  const int n0b = blockIdx.x * 64;
  const int r = t >> 1, part = t & 1;
  const _Float16* ap[2];
  int as_[2];
#pragma unroll
  for (int c = 0; c < 2; ++c) {
    const int gr = row0 + c * 64 + r;
    const int w = gr >> 6, rl = gr & 63;
    if (rl < 49) {
      const int b = w >> 6, wc = w & 63, wi = wc >> 3, wj = wc & 7;
      const int i = rl / 7, j = rl - i * 7;
      const int gh = (wi * 7 + i + 3) % 56, gw = (wj * 7 + j + 3) % 56;
      ap[c] = xn + ((size_t)b * 3136 + gh * 56 + gw) * 384 + part * 16;
      as_[c] = 1;
    } else {
      ap[c] = zbuf + part * 16;
      as_[c] = 0;
    }
  }
  const _Float16* bp = wT + (size_t)(n0b + r) * 384 + part * 16;
  v8f accLo[4], accHi[4];
#pragma unroll
  for (int j = 0; j < 4; ++j) { accLo[j] = vzero8(); accHi[j] = vzero8(); }
  gemm_core_128x64(ap[0], as_[0], ap[1], as_[1], bp, 384, sA, sB, accLo, accHi, t);
  const int h = (lane >> 4) & 1, n15 = lane & 15;
#pragma unroll
  for (int j = 0; j < 4; ++j) {
    const int n = n0b + j * 16 + n15;
    const int which = n / 384, rem = n - which * 384;
    const int head = rem >> 5, dd = rem & 31;
    const float bv = bias[n];
    const float sc = (which == 0) ? 0.17677669529663687f : 1.0f;  // fold 1/sqrt(32) into q
#pragma unroll
    for (int sub = 0; sub < 2; ++sub) {
      const v8f acc = sub ? accHi[j] : accLo[j];
#pragma unroll
      for (int rr = 0; rr < 8; ++rr) {
        const int gr = row0 + wv * 32 + sub * 16 + h * 8 + rr;
        const int w = gr >> 6, rl = gr & 63;
        qkv[(((size_t)w * 3 + which) * 12 + head) * 2048 + (size_t)rl * 32 + dd] =
            (_Float16)((acc[rr] + bv) * sc);
      }
    }
  }
}

// ---------------- fused per-(window,head) attention ----------------
__global__ __launch_bounds__(128) void attn_kernel(const _Float16* __restrict__ qkv,
    const float* __restrict__ rpb, _Float16* __restrict__ att) {
  __shared__ _Float16 sQ[64 * 32];
  __shared__ _Float16 sK[64 * 32];
  __shared__ _Float16 sVT[32 * 64];
  __shared__ float    sS[64 * 64];
  __shared__ _Float16 sP[64 * 64];
  __shared__ float    sRpb[169];
  __shared__ int      sLab[49];
  const int t = threadIdx.x, lane = t & 31, wv = t >> 5;
  const int w = blockIdx.x / 12, head = blockIdx.x - w * 12;
  const int wc = w & 63, wi = wc >> 3, wj = wc & 7;
  const size_t base = (((size_t)w * 3) * 12 + head) * 2048;
  const int r = t >> 1, part = t & 1;
  {
    const uint4* gq = (const uint4*)(qkv + base + r * 32 + part * 16);
    uint4* dq = (uint4*)(sQ + r * 32 + part * 16);
    dq[0] = gq[0]; dq[1] = gq[1];
    const uint4* gk = (const uint4*)(qkv + base + 24576 + r * 32 + part * 16);
    uint4* dk = (uint4*)(sK + r * 32 + part * 16);
    dk[0] = gk[0]; dk[1] = gk[1];
    const uint4* gv = (const uint4*)(qkv + base + 49152 + r * 32 + part * 16);
    const uint4 v0 = gv[0], v1 = gv[1];
    _Float16 tmp[16];
    *(uint4*)(tmp)     = v0;
    *(uint4*)(tmp + 8) = v1;
#pragma unroll
    for (int dd = 0; dd < 16; ++dd) sVT[(part * 16 + dd) * 64 + r] = tmp[dd];
  }
  for (int i = t; i < 169; i += 128) sRpb[i] = rpb[(size_t)i * 12 + head];
  if (t < 49) {
    const int i = t / 7, j = t - i * 7;
    const int hh = wi * 7 + i, ww = wj * 7 + j;
    const int rh = (hh < 49) ? 0 : ((hh < 53) ? 1 : 2);
    const int rw = (ww < 49) ? 0 : ((ww < 53) ? 1 : 2);
    sLab[t] = rh * 3 + rw;
  }
  __syncthreads();
  // S = q @ k^T  (q pre-scaled)
  v8f accS[4];
#pragma unroll
  for (int j = 0; j < 4; ++j) accS[j] = vzero8();
  {
    const v16h a = load_afrag(sQ, wv * 16, 32, 0, lane);
#pragma unroll
    for (int j = 0; j < 4; ++j) {
      const v16h b = load_bfrag(sK, j * 16, 32, 0, lane);
      accS[j] = wmma32(a, b, accS[j]);
    }
  }
  const int h = (lane >> 4) & 1, n15 = lane & 15;
#pragma unroll
  for (int j = 0; j < 4; ++j) {
    const int n = j * 16 + n15;
#pragma unroll
    for (int rr = 0; rr < 8; ++rr) {
      const int m = wv * 16 + h * 8 + rr;
      float val;
      if (n >= 49) val = -1e30f;           // padded key columns -> prob 0
      else if (m >= 49) val = 0.0f;        // padded query rows -> finite, discarded
      else {
        const int mi = m / 7, mj = m - mi * 7, ni = n / 7, nj = n - ni * 7;
        const float bias = sRpb[(mi - ni + 6) * 13 + (mj - nj + 6)];
        const float msk = (sLab[m] != sLab[n]) ? -100.0f : 0.0f;
        val = accS[j][rr] + bias + msk;
      }
      sS[m * 64 + n] = val;
    }
  }
  __syncthreads();
  // softmax: 2 threads per row, pair-combine via lane shuffle
  {
    const int row = t >> 1, half = t & 1;
    float* srow = sS + row * 64 + half * 32;
    _Float16* prow = sP + row * 64 + half * 32;
    float mx = -1e30f;
#pragma unroll
    for (int c = 0; c < 32; ++c) mx = fmaxf(mx, srow[c]);
    mx = fmaxf(mx, __shfl_xor(mx, 1, 32));
    float sum = 0.0f;
#pragma unroll
    for (int c = 0; c < 32; ++c) {
      const float e = __expf(srow[c] - mx);
      sum += e;
      prow[c] = (_Float16)e;
    }
    sum += __shfl_xor(sum, 1, 32);
    const float inv = 1.0f / sum;
#pragma unroll
    for (int c = 0; c < 32; ++c) prow[c] = (_Float16)((float)prow[c] * inv);
  }
  __syncthreads();
  // O = P @ V
  v8f accO[2];
  accO[0] = vzero8(); accO[1] = vzero8();
#pragma unroll
  for (int ks = 0; ks < 2; ++ks) {
    const v16h a = load_afrag(sP, wv * 16, 64, ks * 32, lane);
#pragma unroll
    for (int j = 0; j < 2; ++j) {
      const v16h b = load_bfrag(sVT, j * 16, 64, ks * 32, lane);
      accO[j] = wmma32(a, b, accO[j]);
    }
  }
#pragma unroll
  for (int j = 0; j < 2; ++j) {
    const int dcol = head * 32 + j * 16 + n15;
#pragma unroll
    for (int rr = 0; rr < 8; ++rr) {
      const int row = wv * 16 + h * 8 + rr;
      att[((size_t)w * 64 + row) * 384 + dcol] = (_Float16)accO[j][rr];
    }
  }
}

// ---------------- proj GEMM + window-reverse + roll + residual into d_out ----------------
__global__ __launch_bounds__(128) void gemm_proj_kernel(const _Float16* __restrict__ att,
    const _Float16* __restrict__ wT, const float* __restrict__ bias,
    const float* __restrict__ xin, float* __restrict__ out) {
  __shared__ _Float16 sA[2 * 128 * 32];
  __shared__ _Float16 sB[2 * 64 * 32];
  const int t = threadIdx.x, lane = t & 31, wv = t >> 5;
  const int row0 = blockIdx.y * 128;
  const int n0b = blockIdx.x * 64;
  const int r = t >> 1, part = t & 1;
  const _Float16* a0 = att + (size_t)(row0 + r) * 384 + part * 16;
  const _Float16* a1 = att + (size_t)(row0 + 64 + r) * 384 + part * 16;
  const _Float16* bp = wT + (size_t)(n0b + r) * 384 + part * 16;
  v8f accLo[4], accHi[4];
#pragma unroll
  for (int j = 0; j < 4; ++j) { accLo[j] = vzero8(); accHi[j] = vzero8(); }
  gemm_core_128x64(a0, 1, a1, 1, bp, 384, sA, sB, accLo, accHi, t);
  const int h = (lane >> 4) & 1, n15 = lane & 15;
#pragma unroll
  for (int j = 0; j < 4; ++j) {
    const int n = n0b + j * 16 + n15;
    const float bv = bias[n];
#pragma unroll
    for (int sub = 0; sub < 2; ++sub) {
      const v8f acc = sub ? accHi[j] : accLo[j];
#pragma unroll
      for (int rr = 0; rr < 8; ++rr) {
        const int gr = row0 + wv * 32 + sub * 16 + h * 8 + rr;
        const int w = gr >> 6, rl = gr & 63;
        if (rl < 49) {
          const int b = w >> 6, wc = w & 63, wi = wc >> 3, wj = wc & 7;
          const int i = rl / 7, jj = rl - i * 7;
          const int gh = (wi * 7 + i + 3) % 56, gw = (wj * 7 + jj + 3) % 56;
          const size_t o = ((size_t)b * 3136 + gh * 56 + gw) * 384 + n;
          out[o] = xin[o] + acc[rr] + bv;   // x2 = shortcut + attn_out
        }
      }
    }
  }
}

// ---------------- LN2 followed by ffLN: x2 (f32 in d_out) -> yn (f16) ----------------
__global__ __launch_bounds__(128) void ln2ff_kernel(const float* __restrict__ x2,
    const float* __restrict__ g2, const float* __restrict__ b2,
    const float* __restrict__ gf, const float* __restrict__ bf,
    _Float16* __restrict__ yn) {
  __shared__ float sh[128];
  const size_t tk = blockIdx.x;
  const int t = threadIdx.x;
  const float* xp = x2 + tk * 384;
  const float v0 = xp[t], v1 = xp[t + 128], v2 = xp[t + 256];
  float s  = block_sum128(v0 + v1 + v2, sh, t);
  float s2 = block_sum128(v0 * v0 + v1 * v1 + v2 * v2, sh, t);
  float mu = s * (1.0f / 384.0f);
  float var = s2 * (1.0f / 384.0f) - mu * mu;
  float rs = rsqrtf(var + 1e-5f);
  const float y0 = (v0 - mu) * rs * g2[t]       + b2[t];
  const float y1 = (v1 - mu) * rs * g2[t + 128] + b2[t + 128];
  const float y2 = (v2 - mu) * rs * g2[t + 256] + b2[t + 256];
  s  = block_sum128(y0 + y1 + y2, sh, t);
  s2 = block_sum128(y0 * y0 + y1 * y1 + y2 * y2, sh, t);
  mu = s * (1.0f / 384.0f);
  var = s2 * (1.0f / 384.0f) - mu * mu;
  rs = rsqrtf(var + 1e-6f);
  _Float16* op = yn + tk * 384;
  op[t]       = (_Float16)((y0 - mu) * rs * gf[t]       + bf[t]);
  op[t + 128] = (_Float16)((y1 - mu) * rs * gf[t + 128] + bf[t + 128]);
  op[t + 256] = (_Float16)((y2 - mu) * rs * gf[t + 256] + bf[t + 256]);
}

// ---------------- FF1 GEMM + exact GELU ----------------
__global__ __launch_bounds__(128) void gemm_ff1_kernel(const _Float16* __restrict__ yn,
    const _Float16* __restrict__ wT, const float* __restrict__ bias,
    _Float16* __restrict__ hid) {
  __shared__ _Float16 sA[2 * 128 * 32];
  __shared__ _Float16 sB[2 * 64 * 32];
  const int t = threadIdx.x, lane = t & 31, wv = t >> 5;
  const size_t row0 = (size_t)blockIdx.y * 128;
  const int n0b = blockIdx.x * 64;
  const int r = t >> 1, part = t & 1;
  const _Float16* a0 = yn + (row0 + r) * 384 + part * 16;
  const _Float16* a1 = yn + (row0 + 64 + r) * 384 + part * 16;
  const _Float16* bp = wT + (size_t)(n0b + r) * 384 + part * 16;
  v8f accLo[4], accHi[4];
#pragma unroll
  for (int j = 0; j < 4; ++j) { accLo[j] = vzero8(); accHi[j] = vzero8(); }
  gemm_core_128x64(a0, 1, a1, 1, bp, 384, sA, sB, accLo, accHi, t);
  const int h = (lane >> 4) & 1, n15 = lane & 15;
#pragma unroll
  for (int j = 0; j < 4; ++j) {
    const int n = n0b + j * 16 + n15;
    const float bv = bias[n];
#pragma unroll
    for (int sub = 0; sub < 2; ++sub) {
      const v8f acc = sub ? accHi[j] : accLo[j];
#pragma unroll
      for (int rr = 0; rr < 8; ++rr) {
        const size_t row = row0 + wv * 32 + sub * 16 + h * 8 + rr;
        const float v = acc[rr] + bv;
        const float g = 0.5f * v * (1.0f + erff(v * 0.7071067811865475f));
        hid[row * 1536 + n] = (_Float16)g;
      }
    }
  }
}

// ---------------- FF2 GEMM + final residual: d_out += y ----------------
__global__ __launch_bounds__(128) void gemm_ff2_kernel(const _Float16* __restrict__ hid,
    const _Float16* __restrict__ wT, const float* __restrict__ bias,
    float* __restrict__ out) {
  __shared__ _Float16 sA[2 * 128 * 32];
  __shared__ _Float16 sB[2 * 64 * 32];
  const int t = threadIdx.x, lane = t & 31, wv = t >> 5;
  const size_t row0 = (size_t)blockIdx.y * 128;
  const int n0b = blockIdx.x * 64;
  const int r = t >> 1, part = t & 1;
  const _Float16* a0 = hid + (row0 + r) * 1536 + part * 16;
  const _Float16* a1 = hid + (row0 + 64 + r) * 1536 + part * 16;
  const _Float16* bp = wT + (size_t)(n0b + r) * 1536 + part * 16;
  v8f accLo[4], accHi[4];
#pragma unroll
  for (int j = 0; j < 4; ++j) { accLo[j] = vzero8(); accHi[j] = vzero8(); }
  gemm_core_128x64(a0, 1, a1, 1, bp, 1536, sA, sB, accLo, accHi, t);
  const int h = (lane >> 4) & 1, n15 = lane & 15;
#pragma unroll
  for (int j = 0; j < 4; ++j) {
    const int n = n0b + j * 16 + n15;
    const float bv = bias[n];
#pragma unroll
    for (int sub = 0; sub < 2; ++sub) {
      const v8f acc = sub ? accHi[j] : accLo[j];
#pragma unroll
      for (int rr = 0; rr < 8; ++rr) {
        const size_t row = row0 + wv * 32 + sub * 16 + h * 8 + rr;
        const size_t o = row * 384 + n;
        out[o] = out[o] + acc[rr] + bv;
      }
    }
  }
}

extern "C" void kernel_launch(void* const* d_in, const int* in_sizes, int n_in,
                              void* d_out, int out_size, void* d_ws, size_t ws_size,
                              hipStream_t stream) {
  (void)in_sizes; (void)n_in; (void)out_size; (void)ws_size;
  const float* x      = (const float*)d_in[0];
  const float* ln1_g  = (const float*)d_in[1];
  const float* ln1_b  = (const float*)d_in[2];
  const float* qkv_w  = (const float*)d_in[3];
  const float* qkv_b  = (const float*)d_in[4];
  const float* proj_w = (const float*)d_in[5];
  const float* proj_b = (const float*)d_in[6];
  const float* rpb    = (const float*)d_in[7];
  const float* ln2_g  = (const float*)d_in[8];
  const float* ln2_b  = (const float*)d_in[9];
  const float* ff_g   = (const float*)d_in[10];
  const float* ff_b   = (const float*)d_in[11];
  const float* ff1_w  = (const float*)d_in[12];
  const float* ff1_b  = (const float*)d_in[13];
  const float* ff2_w  = (const float*)d_in[14];
  const float* ff2_b  = (const float*)d_in[15];
  float* out = (float*)d_out;

  char* ws = (char*)d_ws;
  size_t off = 0;
  auto take = [&](size_t bytes) -> char* {
    char* p = ws + off;
    off += (bytes + 255) & ~(size_t)255;
    return p;
  };
  _Float16* wqkvT  = (_Float16*)take(442368ull * 2);     // (1152, 384)
  _Float16* wprojT = (_Float16*)take(147456ull * 2);     // (384, 384)
  _Float16* wff1T  = (_Float16*)take(589824ull * 2);     // (1536, 384)
  _Float16* wff2T  = (_Float16*)take(589824ull * 2);     // (384, 1536)
  _Float16* zbuf   = (_Float16*)take(32ull * 2);         // zero pad row source
  _Float16* xn     = (_Float16*)take(38535168ull * 2);   // (32,3136,384), reused as yn
  _Float16* qkv    = (_Float16*)take(154140672ull * 2);  // q/k/v head-major, reused as hid
  _Float16* att    = (_Float16*)take(50331648ull * 2);   // (2048,64,384)
  _Float16* yn  = xn;
  _Float16* hid = qkv;

  transpose_w_kernel<<<(442368 + 255) / 256, 256, 0, stream>>>(qkv_w, wqkvT, 384, 1152);
  transpose_w_kernel<<<(147456 + 255) / 256, 256, 0, stream>>>(proj_w, wprojT, 384, 384);
  transpose_w_kernel<<<(589824 + 255) / 256, 256, 0, stream>>>(ff1_w, wff1T, 384, 1536);
  transpose_w_kernel<<<(589824 + 255) / 256, 256, 0, stream>>>(ff2_w, wff2T, 1536, 384);
  zero_pad_kernel<<<1, 32, 0, stream>>>(zbuf);

  ln1_kernel<<<100352, 128, 0, stream>>>(x, ln1_g, ln1_b, xn);
  gemm_qkv_kernel<<<dim3(18, 1024), 128, 0, stream>>>(xn, zbuf, wqkvT, qkv_b, qkv);
  attn_kernel<<<24576, 128, 0, stream>>>(qkv, rpb, att);
  gemm_proj_kernel<<<dim3(6, 1024), 128, 0, stream>>>(att, wprojT, proj_b, x, out);
  ln2ff_kernel<<<100352, 128, 0, stream>>>(out, ln2_g, ln2_b, ff_g, ff_b, yn);
  gemm_ff1_kernel<<<dim3(24, 784), 128, 0, stream>>>(yn, wff1T, ff1_b, hid);
  gemm_ff2_kernel<<<dim3(6, 784), 128, 0, stream>>>(hid, wff2T, ff2_b, out);
}
